// ResidualBlockgjz_35442070126788
// MI455X (gfx1250) — matrix operand order repaired and checked by reading, not compiled, not run on hardware
//
#include <hip/hip_runtime.h>
#include <hip/hip_bf16.h>

typedef __attribute__((ext_vector_type(16))) _Float16 v16h;
typedef __attribute__((ext_vector_type(8)))  float    v8f;

#define CCH   96          // channels
#define KT    3           // K tiles of 32 (96/32)
#define NT    6           // N tiles of 16 (96/16)
#define NOFF  26          // non-center offsets
#define NMAT  27          // per layer: Wc + 26 Wk
#define MAT_HALFS (KT*NT*32*16)   // 9216 f16 per packed matrix (18432 B)
#define NBATCH 4

// ---------------------------------------------------------------------------
// Weight prep: fold BN scale into columns, convert to f16, pack into
// WMMA B-fragment layout: frag[(kt*NT+nt)*32 + lane][e] with
//   col = nt*16 + (lane&15),  krow = kt*32 + (lane>=16?8:0) + (e&7) + (e>=8?16:0)
// ---------------------------------------------------------------------------
__global__ void prep_weights_kernel(const float* __restrict__ Wk,
                                    const float* __restrict__ Wc,
                                    const float* __restrict__ gamma,
                                    const float* __restrict__ var,
                                    _Float16* __restrict__ packed)
{
    int mat  = blockIdx.x;            // 0..134
    int j    = mat / NMAT;
    int mloc = mat % NMAT;
    const float* W = (mloc == 0) ? (Wc + (size_t)j * CCH * CCH)
                                 : (Wk + ((size_t)j * NOFF + (mloc - 1)) * CCH * CCH);
    _Float16* dst = packed + (size_t)mat * MAT_HALFS;
    const float* g = gamma + j * CCH;
    const float* v = var   + j * CCH;
    for (int idx = threadIdx.x; idx < MAT_HALFS; idx += blockDim.x) {
        int e    = idx & 15;
        int lane = (idx >> 4) & 31;
        int frag = idx >> 9;          // kt*NT + nt
        int nt   = frag % NT;
        int kt   = frag / NT;
        int col  = nt * 16 + (lane & 15);
        int krow = kt * 32 + ((lane >= 16) ? 8 : 0) + (e & 7) + ((e >= 8) ? 16 : 0);
        float s  = g[col] * rsqrtf(v[col] + 1e-5f);
        dst[idx] = (_Float16)(W[krow * CCH + col] * s);
    }
}

__global__ void prep_bias_kernel(const float* __restrict__ gamma,
                                 const float* __restrict__ beta,
                                 const float* __restrict__ mean,
                                 const float* __restrict__ var,
                                 float* __restrict__ bias)
{
    int i = blockIdx.x * blockDim.x + threadIdx.x;
    if (i < 5 * CCH) {
        float s = gamma[i] * rsqrtf(var[i] + 1e-5f);
        bias[i] = beta[i] - mean[i] * s;
    }
}

// ---------------------------------------------------------------------------
// Inverse neighbor table: nbr[o*NOFF + k] = input row feeding output o at
// offset k, or -1.  Converts the scatter-add conv into a pure gather.
// ---------------------------------------------------------------------------
__global__ void fill_nbr_kernel(int* __restrict__ nbr, int n)
{
    int i = blockIdx.x * blockDim.x + threadIdx.x;
    if (i < n) nbr[i] = -1;
}

__global__ void build_nbr_kernel(const int* __restrict__ in_idx,
                                 const int* __restrict__ out_idx,
                                 const float* __restrict__ pmask,
                                 int* __restrict__ nbr, int M)
{
    int i = blockIdx.x * blockDim.x + threadIdx.x;
    if (i >= NOFF * M) return;
    int k = i / M;
    if (pmask[i] > 0.0f)
        nbr[(size_t)out_idx[i] * NOFF + k] = in_idx[i];
}

// f32x4 -> 4 f16 elements of an A fragment (constant base indices only)
__device__ __forceinline__ void cvt4(v16h& a, int base, float4 f, float mk)
{
    a[base + 0] = (_Float16)(f.x * mk);
    a[base + 1] = (_Float16)(f.y * mk);
    a[base + 2] = (_Float16)(f.z * mk);
    a[base + 3] = (_Float16)(f.w * mk);
}

// A-fragment: row-major f32 row -> 16x32 f16 A layout; K pattern per lane is
// two contiguous runs of 8 at kt*32+(half?8:0) and +16 further (4x float4).
__device__ __forceinline__ v16h load_a_frag(const float* __restrict__ xr,
                                            int kt, int half, float mk)
{
    const float4* p = (const float4*)(xr + kt * 32 + (half ? 8 : 0));
    float4 q0 = p[0], q1 = p[1], q2 = p[4], q3 = p[5];
    v16h a;
    cvt4(a, 0, q0, mk);
    cvt4(a, 4, q1, mk);
    cvt4(a, 8, q2, mk);
    cvt4(a, 12, q3, mk);
    return a;
}

__device__ __forceinline__ void stage_mat(_Float16* __restrict__ sm,
                                          const _Float16* __restrict__ gm)
{
    const uint4* src = (const uint4*)gm;       // 1152 x 16B
    uint4* dst = (uint4*)sm;
    for (int i = threadIdx.x; i < (MAT_HALFS * 2) / 16; i += blockDim.x)
        dst[i] = src[i];
}

// ---------------------------------------------------------------------------
// Fused gather-conv: each wave owns one 16-row output tile and accumulates
// over the center matrix and all 26 neighbor matrices in registers, then one
// non-atomic store with bias (+ optional ReLU).  Weights staged per-block
// into LDS (16 waves amortize).  A gather + f16 convert overlaps the staging
// window; next neighbor index is prefetched one iteration ahead; all 6 B
// fragments of a K-tile are register-preloaded so ds_loads pipeline.
// ---------------------------------------------------------------------------
__global__ void conv_fused_kernel(const float* __restrict__ x,
                                  const _Float16* __restrict__ layerPk,
                                  const float* __restrict__ bias,   // 96
                                  const int* __restrict__ nbr,
                                  float* __restrict__ y,
                                  int N, int doRelu)
{
    __shared__ _Float16 smem[MAT_HALFS];       // 18432 B
    int wavesPerBlock = blockDim.x >> 5;
    int wid  = blockIdx.x * wavesPerBlock + (threadIdx.x >> 5);
    int lane = threadIdx.x & 31;
    int half = lane >> 4;
    int l15  = lane & 15;
    int row0 = wid << 4;
    int arow = row0 + l15;

    v8f zero = {0.f, 0.f, 0.f, 0.f, 0.f, 0.f, 0.f, 0.f};
    v8f acc[NT];
#pragma unroll
    for (int t = 0; t < NT; ++t) acc[t] = zero;

    const v16h* bs = (const v16h*)smem;
    int srcCur = (arow < N) ? arow : -1;       // km==0: center (identity)

    for (int km = 0; km < NMAT; ++km) {
        __syncthreads();                       // prior smem reads complete
        stage_mat(smem, layerPk + (size_t)km * MAT_HALFS);

        // --- overlap with staging: gather A row, convert, prefetch next idx
        float mk = (srcCur >= 0) ? 1.0f : 0.0f;
        const float* xr = x + (size_t)((srcCur >= 0) ? srcCur : 0) * CCH;
        v16h af[KT];
#pragma unroll
        for (int kt = 0; kt < KT; ++kt)
            af[kt] = load_a_frag(xr, kt, half, mk);
        int srcNext = (km < NMAT - 1 && arow < N)
                        ? nbr[(size_t)arow * NOFF + km] : -1;

        __syncthreads();                       // staged weights visible

#pragma unroll
        for (int kt = 0; kt < KT; ++kt) {
            v16h b[NT];
#pragma unroll
            for (int nt = 0; nt < NT; ++nt)
                b[nt] = bs[(kt * NT + nt) * 32 + lane];
#pragma unroll
            for (int nt = 0; nt < NT; ++nt)
                acc[nt] = __builtin_amdgcn_wmma_f32_16x16x32_f16(
                    false, af[kt], false, b[nt], (short)0, acc[nt], false, false);
        }
        srcCur = srcNext;
    }

#pragma unroll
    for (int nt = 0; nt < NT; ++nt) {
        int col = nt * 16 + l15;
        float bv = bias[col];
#pragma unroll
        for (int r = 0; r < 8; ++r) {
            int row = row0 + r + (half ? 8 : 0);
            if (row < N) {
                float v = acc[nt][r] + bv;
                if (doRelu) v = fmaxf(v, 0.f);
                y[(size_t)row * CCH + col] = v;
            }
        }
    }
}

__global__ void zero_kernel(float* __restrict__ p, int n)
{
    int i = blockIdx.x * blockDim.x + threadIdx.x;
    if (i < n) p[i] = 0.f;
}

// ---------------------------------------------------------------------------
// Per-batch avg-sum + max pooling over k1|k2|k3 (batches are contiguous rows).
// maxs init 0 is valid: inputs are post-ReLU (>= 0).
// ---------------------------------------------------------------------------
__global__ void pool_kernel(const float* __restrict__ k1,
                            const float* __restrict__ k2,
                            const float* __restrict__ k3,
                            float* __restrict__ sums,   // [B][3C]
                            float* __restrict__ maxs,   // [B][3C]
                            int NPB, int splits)
{
    int blk = blockIdx.x;                 // b*3*splits + s*splits + sp
    int sp  = blk % splits;
    int s   = (blk / splits) % 3;
    int b   = blk / (3 * splits);
    const float* src = (s == 0) ? k1 : (s == 1) ? k2 : k3;
    int chunk = (NPB + splits - 1) / splits;
    int r0 = sp * chunk;
    int r1 = r0 + chunk; if (r1 > NPB) r1 = NPB;
    int c  = threadIdx.x;
    if (c >= CCH) return;
    float sum = 0.f, mx = 0.f;
    for (int r = r0; r < r1; ++r) {
        float v = src[(size_t)(b * NPB + r) * CCH + c];
        sum += v;
        mx = fmaxf(mx, v);
    }
    int o = b * (3 * CCH) + s * CCH + c;
    atomicAdd(&sums[o], sum);
    atomicMax((int*)&maxs[o], __float_as_int(mx));
}

// ---------------------------------------------------------------------------
// Tiny attention MLP: att = sigmoid(fc(avg) + fc(max)); wsig = sigmoid(nx_w)
// ---------------------------------------------------------------------------
__global__ void att_kernel(const float* __restrict__ sums,
                           const float* __restrict__ maxs,
                           const float* __restrict__ fc1_w,
                           const float* __restrict__ fc1_b,
                           const float* __restrict__ fc2_w,
                           const float* __restrict__ fc2_b,
                           const float* __restrict__ nx_w,
                           float* __restrict__ att,
                           float* __restrict__ wsig,
                           int NPB, int H, int R)
{
    __shared__ float sh[2 * NBATCH * 64];   // hidden for avg (0) and max (1)
    int tid = threadIdx.x;
    for (int i = tid; i < 2 * NBATCH * R; i += blockDim.x) {
        int which = i / (NBATCH * R);
        int b = (i / R) % NBATCH;
        int r = i % R;
        const float* z = (which == 0) ? sums : maxs;
        float accv = fc1_b[r];
        float inv = 1.0f / (float)NPB;
        for (int h = 0; h < H; ++h) {
            float zv = z[b * H + h];
            if (which == 0) zv *= inv;
            accv += zv * fc1_w[h * R + r];
        }
        sh[which * NBATCH * R + b * R + r] = fmaxf(accv, 0.f);
    }
    __syncthreads();
    for (int i = tid; i < NBATCH * H; i += blockDim.x) {
        int b = i / H, h = i % H;
        float accv = 2.f * fc2_b[h];
        for (int r = 0; r < R; ++r)
            accv += (sh[b * R + r] + sh[NBATCH * R + b * R + r]) * fc2_w[r * H + h];
        att[i] = 1.f / (1.f + __expf(-accv));
    }
    if (tid < 3) wsig[tid] = 1.f / (1.f + __expf(-nx_w[tid]));
}

// ---------------------------------------------------------------------------
// Final combine: out = relu( sum_s (k_s + xcat[3c+s]*att[3c+s]) * w_s + feats )
// ---------------------------------------------------------------------------
__global__ void final_kernel(const float* __restrict__ feats,
                             const float* __restrict__ k1,
                             const float* __restrict__ k2,
                             const float* __restrict__ k3,
                             const float* __restrict__ att,
                             const float* __restrict__ wsig,
                             const int* __restrict__ bidx,
                             float* __restrict__ out, int N)
{
    int i = blockIdx.x * blockDim.x + threadIdx.x;
    if (i >= N * CCH) return;
    int n = i / CCH, c = i % CCH;
    int b = bidx[n];
    float w0 = wsig[0], w1 = wsig[1], w2 = wsig[2];
    float o = 0.f;
#pragma unroll
    for (int s = 0; s < 3; ++s) {
        int jj = 3 * c + s;
        int which = jj / CCH;
        int cc = jj - which * CCH;
        const float* xp = (which == 0) ? k1 : (which == 1) ? k2 : k3;
        const float* kp = (s == 0) ? k1 : (s == 1) ? k2 : k3;
        float xv = xp[(size_t)n * CCH + cc];
        float kv = kp[(size_t)n * CCH + c];
        float ws = (s == 0) ? w0 : (s == 1) ? w1 : w2;
        o += (kv + xv * att[b * (3 * CCH) + jj]) * ws;
    }
    out[i] = fmaxf(o + feats[i], 0.f);
}

// ---------------------------------------------------------------------------
extern "C" void kernel_launch(void* const* d_in, const int* in_sizes, int n_in,
                              void* d_out, int out_size, void* d_ws, size_t ws_size,
                              hipStream_t stream)
{
    const float* feats   = (const float*)d_in[0];
    const float* Wk      = (const float*)d_in[1];
    const float* Wc      = (const float*)d_in[2];
    const float* gamma   = (const float*)d_in[3];
    const float* beta    = (const float*)d_in[4];
    const float* mean    = (const float*)d_in[5];
    const float* var     = (const float*)d_in[6];
    const float* fc1_w   = (const float*)d_in[7];
    const float* fc1_b   = (const float*)d_in[8];
    const float* fc2_w   = (const float*)d_in[9];
    const float* fc2_b   = (const float*)d_in[10];
    const float* nx_w    = (const float*)d_in[11];
    const float* pmask   = (const float*)d_in[12];
    const int*   in_idx  = (const int*)d_in[13];
    const int*   out_idx = (const int*)d_in[14];
    const int*   bidx    = (const int*)d_in[15];

    const int N   = in_sizes[0] / CCH;
    const int M   = in_sizes[12] / NOFF;
    const int H   = in_sizes[10];   // 3*C = 288
    const int R   = in_sizes[8];    // 18
    const int NPB = N / NBATCH;

    // workspace bump allocator (256B aligned)
    char* ws = (char*)d_ws;
    size_t off = 0;
    auto alloc = [&](size_t bytes) -> void* {
        void* p = ws + off;
        off += (bytes + 255) & ~(size_t)255;
        return p;
    };
    _Float16* packed = (_Float16*)alloc((size_t)5 * NMAT * MAT_HALFS * 2);
    float* bias = (float*)alloc(5 * CCH * 4);
    float* sums = (float*)alloc((size_t)NBATCH * H * 4);
    float* maxs = (float*)alloc((size_t)NBATCH * H * 4);
    float* attb = (float*)alloc((size_t)NBATCH * H * 4);
    float* wsig = (float*)alloc(64);
    int*   nbr  = (int*)alloc((size_t)N * NOFF * 4);
    size_t bufBytes = (size_t)N * CCH * 4;
    float* bufA = (float*)alloc(bufBytes);
    float* bufB = (float*)alloc(bufBytes);
    float* bufC = (float*)alloc(bufBytes);
    (void)ws_size; (void)n_in; (void)out_size;

    // 1) prep folded/packed weights + bias + inverse neighbor table
    prep_weights_kernel<<<5 * NMAT, 256, 0, stream>>>(Wk, Wc, gamma, var, packed);
    prep_bias_kernel<<<(5 * CCH + 255) / 256, 256, 0, stream>>>(gamma, beta, mean, var, bias);
    fill_nbr_kernel<<<(N * NOFF + 255) / 256, 256, 0, stream>>>(nbr, N * NOFF);
    build_nbr_kernel<<<(NOFF * M + 255) / 256, 256, 0, stream>>>(in_idx, out_idx, pmask, nbr, M);

    // 2) five fused gather-conv layers (no atomics, single store each)
    const int rowTiles = (N + 15) >> 4;
    const int wavesPerBlock = 16;                       // 512 threads
    const int convBlocks = (rowTiles + wavesPerBlock - 1) / wavesPerBlock;

    const float* xs[5] = {feats, bufA, bufB, bufC, bufA};
    float*       ys[5] = {bufA,  bufB, bufC, bufA, bufB};
    const int    rl[5] = {1, 0, 1, 1, 1};

    for (int j = 0; j < 5; ++j) {
        const _Float16* layerPk = packed + (size_t)j * NMAT * MAT_HALFS;
        conv_fused_kernel<<<convBlocks, wavesPerBlock * 32, 0, stream>>>(
            xs[j], layerPk, bias + j * CCH, nbr, ys[j], N, rl[j]);
    }
    // k1 = bufC, k2 = bufA, k3 = bufB

    // 3) pooling + attention
    zero_kernel<<<(NBATCH * H + 255) / 256, 256, 0, stream>>>(sums, NBATCH * H);
    zero_kernel<<<(NBATCH * H + 255) / 256, 256, 0, stream>>>(maxs, NBATCH * H);
    const int splits = 64;
    pool_kernel<<<NBATCH * 3 * splits, 96, 0, stream>>>(bufC, bufA, bufB, sums, maxs, NPB, splits);
    att_kernel<<<1, 320, 0, stream>>>(sums, maxs, fc1_w, fc1_b, fc2_w, fc2_b, nx_w,
                                      attb, wsig, NPB, H, R);

    // 4) final combine + residual
    final_kernel<<<((N * CCH) + 255) / 256, 256, 0, stream>>>(
        feats, bufC, bufA, bufB, attb, wsig, bidx, (float*)d_out, N);
}
